// MultiHeadAttention_39522289058125
// MI455X (gfx1250) — matrix operand order repaired and checked by reading, not compiled, not run on hardware
//
#include <hip/hip_runtime.h>
#include <hip/hip_bf16.h>

// ---------------- problem constants ----------------
#define B_   32
#define S_   512
#define D_   512
#define H_   8
#define HD_  64
#define CB_  8                 // batch chunk for attention score buffer
#define NCHUNK (B_ / CB_)

typedef __attribute__((ext_vector_type(16))) __bf16 v16bf;
typedef __attribute__((ext_vector_type(8)))  float  v8f;

union BF16x16 { v16bf v; unsigned int u[8]; };

__device__ __forceinline__ float b2f(unsigned short h) {
  unsigned int u = ((unsigned int)h) << 16;
  float f; __builtin_memcpy(&f, &u, 4); return f;
}
__device__ __forceinline__ unsigned short f2b(float f) {
  unsigned int u; __builtin_memcpy(&u, &f, 4);
  u += 0x7FFFu + ((u >> 16) & 1u);          // round-to-nearest-even
  return (unsigned short)(u >> 16);
}

// ---------------- generic batched bf16 WMMA GEMM ----------------
// C[z] = alpha * A[z](MxK) * B[z](KxN, or NxK if TRANSB) + R   (R optional, f32)
// per-z offsets: off = (z/div)*s0 + (z%div)*s1
struct GemmP {
  const unsigned short* A; const unsigned short* Bm;
  float* Cf; unsigned short* Cb; const float* R;
  long sA0, sA1, sB0, sB1, sC0, sC1, sR0;
  int  divA, divB, divC;
  int  lda, ldb, ldc, ldr;
  int  K; float alpha;
};

#define LDA_S 40   // u16 stride for A tile: 80B rows -> every 16B lane-chunk 16B-aligned,
                   // frag reads at dword 20*m (+k/2) stay spread across banks
#define LDB_P 68   // dword stride for pair-packed B tile (half-wave disjoint banks)

template <int TRANSB>
__global__ __launch_bounds__(256) void gemm_wmma(GemmP p) {
  __shared__ __attribute__((aligned(16))) unsigned short As[64 * LDA_S];
  __shared__ __attribute__((aligned(16))) unsigned int   Bp[16 * LDB_P];
  // Bp[kp][n] = B[2kp][n] | B[2kp+1][n]<<16

  const int z = blockIdx.z;
  const unsigned short* A  = p.A  + (long)(z / p.divA) * p.sA0 + (long)(z % p.divA) * p.sA1;
  const unsigned short* Bm = p.Bm + (long)(z / p.divB) * p.sB0 + (long)(z % p.divB) * p.sB1;

  const int tid  = threadIdx.x;
  const int lane = tid & 31, wave = tid >> 5;
  const int wr = wave >> 1, wc = wave & 1;        // 4x2 wave grid -> 64x64 C tile
  const int m0 = blockIdx.x * 64, n0 = blockIdx.y * 64;

  v8f acc0 = {}; v8f acc1 = {};

  const int ar = tid >> 2, ac = (tid & 3) * 8;    // A tile loader: 64 rows x 32 k
  const int kp = tid >> 4, nc = (tid & 15) * 4;   // B loader (KxN): pair-row kp, 4 cols
  const int tn = tid >> 2, tk = (tid & 3) * 8;    // B loader (NxK, TRANSB)

  // per-lane LDS destination (byte address) for async A staging: 16B-aligned
  const unsigned int lds_a = (unsigned int)(unsigned long long)&As[ar * LDA_S + ac];

  for (int k0 = 0; k0 < p.K; k0 += 32) {
    // ---- stage A tile via CDNA5 async global->LDS DMA path (ASYNCcnt) ----
    {
      const unsigned long long ga =
          (unsigned long long)(A + (long)(m0 + ar) * p.lda + k0 + ac);
      asm volatile("global_load_async_to_lds_b128 %0, %1, off"
                   :: "v"(lds_a), "v"(ga) : "memory");
    }

    // ---- stage B tile pair-packed into dwords (synchronous: needs packing/scatter) ----
    if (TRANSB) {
      // source rows are N, contiguous along K: each dword already holds (k,k+1)
      const uint4 d = *(const uint4*)(Bm + (long)(n0 + tn) * p.ldb + k0 + tk);
      const int kpb = tk >> 1;
      Bp[(kpb + 0) * LDB_P + tn] = d.x;
      Bp[(kpb + 1) * LDB_P + tn] = d.y;
      Bp[(kpb + 2) * LDB_P + tn] = d.z;
      Bp[(kpb + 3) * LDB_P + tn] = d.w;
      __builtin_prefetch(Bm + (long)(n0 + tn) * p.ldb + k0 + 32 + tk, 0, 1);
    } else {
      // source row-major KxN: load rows 2kp and 2kp+1, interleave into packed dwords
      const uint2 lo = *(const uint2*)(Bm + (long)(k0 + 2 * kp)     * p.ldb + n0 + nc);
      const uint2 hi = *(const uint2*)(Bm + (long)(k0 + 2 * kp + 1) * p.ldb + n0 + nc);
      const unsigned short* ls = (const unsigned short*)&lo;
      const unsigned short* hs = (const unsigned short*)&hi;
      uint4 pk;
      pk.x = (unsigned int)ls[0] | ((unsigned int)hs[0] << 16);
      pk.y = (unsigned int)ls[1] | ((unsigned int)hs[1] << 16);
      pk.z = (unsigned int)ls[2] | ((unsigned int)hs[2] << 16);
      pk.w = (unsigned int)ls[3] | ((unsigned int)hs[3] << 16);
      *(uint4*)&Bp[kp * LDB_P + nc] = pk;   // 16B aligned: (68*kp + nc) % 4 == 0
      __builtin_prefetch(Bm + (long)(k0 + 32 + 2 * kp) * p.ldb + n0 + nc, 0, 1);
    }
    // all of this wave's async LDS writes must land before the barrier
    asm volatile("s_wait_asynccnt 0x0" ::: "memory");
    __syncthreads();

    // ---- A fragment (contiguous k order: k = 16*half + s, matching B) ----
    BF16x16 a, b0, b1;
    {
      const int ml = wr * 16 + (lane & 15);
      const int half = lane >> 4;
      #pragma unroll
      for (int i = 0; i < 8; ++i) {
        const int kb = half * 16 + 2 * i;
        a.u[i] = *(const unsigned int*)&As[ml * LDA_S + kb];
      }
    }
    // ---- B fragments: packed-dword reads; fill b0 fully, then b1, so each
    //      fragment's 8 dwords can coalesce into contiguous register tuples ----
    {
      const int nl  = wc * 32 + (lane & 15);
      const int kpb = (lane >> 4) * 8;      // dword pair-row base per half-wave
      #pragma unroll
      for (int i = 0; i < 8; ++i)
        b0.u[i] = Bp[(kpb + i) * LDB_P + nl];
      #pragma unroll
      for (int i = 0; i < 8; ++i)
        b1.u[i] = Bp[(kpb + i) * LDB_P + nl + 16];
    }
    acc0 = __builtin_amdgcn_wmma_f32_16x16x32_bf16(false, a.v, false, b0.v, (short)0, acc0, false, false);
    acc1 = __builtin_amdgcn_wmma_f32_16x16x32_bf16(false, a.v, false, b1.v, (short)0, acc1, false, false);
    __syncthreads();
  }

  // ---- epilogue: scale, optional residual, f32 or bf16 store ----
  const long coff = (long)(z / p.divC) * p.sC0 + (long)(z % p.divC) * p.sC1;
  const int nl = lane & 15, mh = (lane >> 4) * 8;
  #pragma unroll
  for (int i = 0; i < 8; ++i) {
    const int gm  = m0 + wr * 16 + mh + i;
    const int gn0 = n0 + wc * 32 + nl;
    float v0 = acc0[i] * p.alpha;
    float v1 = acc1[i] * p.alpha;
    if (p.R) {
      const float* Rz = p.R + (long)z * p.sR0;
      v0 += Rz[(long)gm * p.ldr + gn0];
      v1 += Rz[(long)gm * p.ldr + gn0 + 16];
    }
    if (p.Cf) {
      float* C = p.Cf + coff;
      C[(long)gm * p.ldc + gn0]      = v0;
      C[(long)gm * p.ldc + gn0 + 16] = v1;
    } else {
      unsigned short* C = p.Cb + coff;
      C[(long)gm * p.ldc + gn0]      = f2b(v0);
      C[(long)gm * p.ldc + gn0 + 16] = f2b(v1);
    }
  }
}

// ---------------- transpose + f32->bf16 convert: out[z][c][r] = in[z][r][c] ----------------
__global__ __launch_bounds__(256) void tconv_f2b(const float* in, unsigned short* out,
                                                 int Rr, int Cc) {
  __shared__ float tile[32][33];
  const int z = blockIdx.z;
  const long base = (long)z * Rr * Cc;
  const int c0 = blockIdx.x * 32, r0 = blockIdx.y * 32;
  const int tx = threadIdx.x & 31, ty = threadIdx.x >> 5;
  #pragma unroll
  for (int j = 0; j < 32; j += 8)
    tile[ty + j][tx] = in[base + (long)(r0 + ty + j) * Cc + c0 + tx];
  __syncthreads();
  #pragma unroll
  for (int j = 0; j < 32; j += 8)
    out[base + (long)(c0 + ty + j) * Rr + r0 + tx] = f2b(tile[tx][ty + j]);
}

// ---------------- out_inter: f1[b][s][d] = Y[b][d][s] + x[b][s][d]; also bf16 copy ----------------
__global__ __launch_bounds__(256) void transadd(const float* Y, const float* x,
                                                float* outf, unsigned short* outb) {
  __shared__ float tile[32][33];
  const int z = blockIdx.z;
  const long base = (long)z * S_ * D_;
  const int c0 = blockIdx.x * 32;   // s block
  const int r0 = blockIdx.y * 32;   // d block
  const int tx = threadIdx.x & 31, ty = threadIdx.x >> 5;
  #pragma unroll
  for (int j = 0; j < 32; j += 8)
    tile[ty + j][tx] = Y[base + (long)(r0 + ty + j) * S_ + c0 + tx];
  __syncthreads();
  #pragma unroll
  for (int j = 0; j < 32; j += 8) {
    const long o = base + (long)(c0 + ty + j) * D_ + r0 + tx;
    const float v = tile[tx][ty + j] + x[o];
    outf[o] = v;
    outb[o] = f2b(v);
  }
}

// ---------------- row softmax over 512, bf16 in/out, in place ----------------
__global__ __launch_bounds__(256) void softmax_rows(unsigned short* Sm) {
  const long row = blockIdx.x;
  unsigned short* p = Sm + row * 512;
  const int t = threadIdx.x;
  const float v0 = b2f(p[t]), v1 = b2f(p[t + 256]);
  __shared__ float red[256];
  red[t] = fmaxf(v0, v1); __syncthreads();
  #pragma unroll
  for (int s = 128; s > 0; s >>= 1) { if (t < s) red[t] = fmaxf(red[t], red[t + s]); __syncthreads(); }
  const float mx = red[0]; __syncthreads();
  const float e0 = __expf(v0 - mx), e1 = __expf(v1 - mx);
  red[t] = e0 + e1; __syncthreads();
  #pragma unroll
  for (int s = 128; s > 0; s >>= 1) { if (t < s) red[t] += red[t + s]; __syncthreads(); }
  const float inv = 1.0f / red[0];
  p[t] = f2b(e0 * inv); p[t + 256] = f2b(e1 * inv);
}

// ---------------- final gated blend ----------------
__global__ __launch_bounds__(256) void blend_k(float* out, const float* inter,
                                               const float* alpha, long n) {
  const long i = (long)blockIdx.x * 256 + threadIdx.x;
  if (i < n) {
    const float a = 1.0f / (1.0f + __expf(-alpha[0]));
    out[i] = (1.0f - a) * out[i] + a * inter[i];
  }
}

// ---------------- host orchestration ----------------
extern "C" void kernel_launch(void* const* d_in, const int* in_sizes, int n_in,
                              void* d_out, int out_size, void* d_ws, size_t ws_size,
                              hipStream_t stream) {
  (void)in_sizes; (void)n_in; (void)out_size; (void)ws_size;
  const float* x     = (const float*)d_in[0];
  const float* mask  = (const float*)d_in[1];
  const float* WqI   = (const float*)d_in[2];
  const float* WkI   = (const float*)d_in[3];
  const float* WvI   = (const float*)d_in[4];
  const float* Wqi   = (const float*)d_in[5];
  const float* Wki   = (const float*)d_in[6];
  const float* Wvi   = (const float*)d_in[7];
  const float* Wpi   = (const float*)d_in[8];
  const float* Wpo   = (const float*)d_in[9];
  const float* WsI   = (const float*)d_in[10];
  const float* WoI   = (const float*)d_in[11];
  const float* Wsi   = (const float*)d_in[12];
  const float* Woi   = (const float*)d_in[13];
  const float* alpha = (const float*)d_in[14];
  float* out = (float*)d_out;

  const long ACT  = (long)B_ * S_ * D_;          // 8388608
  const long MAT  = (long)S_ * D_;               // 262144
  const long HW   = (long)H_ * HD_ * HD_;        // 32768
  const long QKV  = (long)B_ * H_ * S_ * HD_;    // per (b,h): 32768
  const long SIMS = (long)CB_ * H_ * S_ * S_;    // chunked scores

  char* w = (char*)d_ws;
  auto alloc = [&](long elems, int esz) -> void* {
    void* p = (void*)w;
    w += ((long)elems * esz + 255) & ~255L;
    return p;
  };
  unsigned short* bf0 = (unsigned short*)alloc(ACT, 2);
  unsigned short* bf1 = (unsigned short*)alloc(ACT, 2);
  unsigned short* qb  = (unsigned short*)alloc(QKV, 2);
  unsigned short* kb  = (unsigned short*)alloc(QKV, 2);
  unsigned short* vb  = (unsigned short*)alloc(QKV, 2);
  unsigned short* sim = (unsigned short*)alloc(SIMS, 2);
  float* f0 = (float*)alloc(ACT, 4);
  float* f1 = (float*)alloc(ACT, 4);
  unsigned short* wpiT = (unsigned short*)alloc(MAT, 2);
  unsigned short* wpoT = (unsigned short*)alloc(MAT, 2);
  unsigned short* wsIT = (unsigned short*)alloc(MAT, 2);
  unsigned short* woIT = (unsigned short*)alloc(MAT, 2);
  unsigned short* wsiT = (unsigned short*)alloc(MAT, 2);
  unsigned short* woiT = (unsigned short*)alloc(MAT, 2);
  unsigned short* wqIT = (unsigned short*)alloc(HW, 2);
  unsigned short* wkIT = (unsigned short*)alloc(HW, 2);
  unsigned short* wvIT = (unsigned short*)alloc(HW, 2);
  unsigned short* wqiT = (unsigned short*)alloc(HW, 2);
  unsigned short* wkiT = (unsigned short*)alloc(HW, 2);
  unsigned short* wviT = (unsigned short*)alloc(HW, 2);

  const dim3 t256(256);
  // weight transposes (implements "@ W.T" as row-major B = W^T, bf16)
  tconv_f2b<<<dim3(16, 16, 1), t256, 0, stream>>>(Wpi, wpiT, 512, 512);
  tconv_f2b<<<dim3(16, 16, 1), t256, 0, stream>>>(Wpo, wpoT, 512, 512);
  tconv_f2b<<<dim3(16, 16, 1), t256, 0, stream>>>(WsI, wsIT, 512, 512);
  tconv_f2b<<<dim3(16, 16, 1), t256, 0, stream>>>(WoI, woIT, 512, 512);
  tconv_f2b<<<dim3(16, 16, 1), t256, 0, stream>>>(Wsi, wsiT, 512, 512);
  tconv_f2b<<<dim3(16, 16, 1), t256, 0, stream>>>(Woi, woiT, 512, 512);
  tconv_f2b<<<dim3(2, 2, H_), t256, 0, stream>>>(WqI, wqIT, 64, 64);
  tconv_f2b<<<dim3(2, 2, H_), t256, 0, stream>>>(WkI, wkIT, 64, 64);
  tconv_f2b<<<dim3(2, 2, H_), t256, 0, stream>>>(WvI, wvIT, 64, 64);
  tconv_f2b<<<dim3(2, 2, H_), t256, 0, stream>>>(Wqi, wqiT, 64, 64);
  tconv_f2b<<<dim3(2, 2, H_), t256, 0, stream>>>(Wki, wkiT, 64, 64);
  tconv_f2b<<<dim3(2, 2, H_), t256, 0, stream>>>(Wvi, wviT, 64, 64);
  // x^T -> bf0 [B][D][S]
  tconv_f2b<<<dim3(16, 16, B_), t256, 0, stream>>>(x, bf0, 512, 512);

  auto G = [&](const unsigned short* A, int lda, long sA0, long sA1, int divA,
               const unsigned short* Bm, int ldb, long sB0, long sB1, int divB, int transB,
               float* Cf, unsigned short* Cb, int ldc, long sC0, long sC1, int divC,
               const float* R, int ldr, long sR0,
               int M, int N, int K, float al, int Z) {
    GemmP p;
    p.A = A; p.Bm = Bm; p.Cf = Cf; p.Cb = Cb; p.R = R;
    p.sA0 = sA0; p.sA1 = sA1; p.sB0 = sB0; p.sB1 = sB1;
    p.sC0 = sC0; p.sC1 = sC1; p.sR0 = sR0;
    p.divA = divA; p.divB = divB; p.divC = divC;
    p.lda = lda; p.ldb = ldb; p.ldc = ldc; p.ldr = ldr;
    p.K = K; p.alpha = al;
    if (transB)
      gemm_wmma<1><<<dim3(M / 64, N / 64, Z), dim3(256), 0, stream>>>(p);
    else
      gemm_wmma<0><<<dim3(M / 64, N / 64, Z), dim3(256), 0, stream>>>(p);
  };

  // ===== inter path =====
  // xp1 = x^T @ Wpi^T : bf0 -> bf1
  G(bf0, 512, MAT, 0, 1,  wpiT, 512, 0, 0, 1, 0,  nullptr, bf1, 512, MAT, 0, 1,
    nullptr, 0, 0,  512, 512, 512, 1.0f, B_);
  // xp = xp1 @ WsplitInter^T : bf1 -> bf0
  G(bf1, 512, MAT, 0, 1,  wsIT, 512, 0, 0, 1, 0,  nullptr, bf0, 512, MAT, 0, 1,
    nullptr, 0, 0,  512, 512, 512, 1.0f, B_);
  // per-head q/k/v  (z = b*H + h; A = xp col-slice h*64; C = [b][h][512][64])
  G(bf0, 512, MAT, 64, H_,  wqIT, 64, 0, 4096, H_, 0,  nullptr, qb, 64, 32768, 0, 1,
    nullptr, 0, 0,  512, 64, 64, 1.0f, B_ * H_);
  G(bf0, 512, MAT, 64, H_,  wkIT, 64, 0, 4096, H_, 0,  nullptr, kb, 64, 32768, 0, 1,
    nullptr, 0, 0,  512, 64, 64, 1.0f, B_ * H_);
  G(bf0, 512, MAT, 64, H_,  wvIT, 64, 0, 4096, H_, 0,  nullptr, vb, 64, 32768, 0, 1,
    nullptr, 0, 0,  512, 64, 64, 1.0f, B_ * H_);
  // attention (no mask), concat into bf1
  for (int cb = 0; cb < NCHUNK; ++cb) {
    const long qoff = (long)cb * CB_ * H_ * 32768;
    G(qb + qoff, 64, 32768, 0, 1,  kb + qoff, 64, 32768, 0, 1, 1 /*transB*/,
      nullptr, sim, 512, (long)S_ * S_, 0, 1,  nullptr, 0, 0,
      512, 512, 64, 0.125f, CB_ * H_);
    softmax_rows<<<dim3(CB_ * H_ * S_), t256, 0, stream>>>(sim);
    G(sim, 512, (long)S_ * S_, 0, 1,  vb + qoff, 64, 32768, 0, 1, 0,
      nullptr, bf1 + (long)cb * CB_ * MAT, 512, MAT, 64, H_,  nullptr, 0, 0,
      512, 64, 512, 1.0f, CB_ * H_);
  }
  // concat @ WoutInter^T : bf1 -> bf0
  G(bf1, 512, MAT, 0, 1,  woIT, 512, 0, 0, 1, 0,  nullptr, bf0, 512, MAT, 0, 1,
    nullptr, 0, 0,  512, 512, 512, 1.0f, B_);
  // @ Wpo^T -> f0 (f32, [B][D][S])
  G(bf0, 512, MAT, 0, 1,  wpoT, 512, 0, 0, 1, 0,  f0, nullptr, 512, MAT, 0, 1,
    nullptr, 0, 0,  512, 512, 512, 1.0f, B_);
  // out_inter = transpose + residual -> f1 (f32) and bf0 (bf16)
  transadd<<<dim3(16, 16, B_), t256, 0, stream>>>(f0, x, f1, bf0);

  // ===== intra path =====
  // xi = out_inter @ WsplitIntra^T : bf0 -> bf1
  G(bf0, 512, MAT, 0, 1,  wsiT, 512, 0, 0, 1, 0,  nullptr, bf1, 512, MAT, 0, 1,
    nullptr, 0, 0,  512, 512, 512, 1.0f, B_);
  // per-head q/k/v intra
  G(bf1, 512, MAT, 64, H_,  wqiT, 64, 0, 4096, H_, 0,  nullptr, qb, 64, 32768, 0, 1,
    nullptr, 0, 0,  512, 64, 64, 1.0f, B_ * H_);
  G(bf1, 512, MAT, 64, H_,  wkiT, 64, 0, 4096, H_, 0,  nullptr, kb, 64, 32768, 0, 1,
    nullptr, 0, 0,  512, 64, 64, 1.0f, B_ * H_);
  G(bf1, 512, MAT, 64, H_,  wviT, 64, 0, 4096, H_, 0,  nullptr, vb, 64, 32768, 0, 1,
    nullptr, 0, 0,  512, 64, 64, 1.0f, B_ * H_);
  // attention with mask residual, concat into bf0
  for (int cb = 0; cb < NCHUNK; ++cb) {
    const long qoff = (long)cb * CB_ * H_ * 32768;
    G(qb + qoff, 64, 32768, 0, 1,  kb + qoff, 64, 32768, 0, 1, 1 /*transB*/,
      nullptr, sim, 512, (long)S_ * S_, 0, 1,  mask, 512, 0 /*broadcast*/,
      512, 512, 64, 0.125f, CB_ * H_);
    softmax_rows<<<dim3(CB_ * H_ * S_), t256, 0, stream>>>(sim);
    G(sim, 512, (long)S_ * S_, 0, 1,  vb + qoff, 64, 32768, 0, 1, 0,
      nullptr, bf0 + (long)cb * CB_ * MAT, 512, MAT, 64, H_,  nullptr, 0, 0,
      512, 64, 512, 1.0f, CB_ * H_);
  }
  // out = concat @ WoutIntra^T -> d_out (f32)
  G(bf0, 512, MAT, 0, 1,  woiT, 512, 0, 0, 1, 0,  out, nullptr, 512, MAT, 0, 1,
    nullptr, 0, 0,  512, 512, 512, 1.0f, B_);
  // blend: out = (1-sigmoid(alpha))*out + sigmoid(alpha)*out_inter
  blend_k<<<dim3((unsigned)((ACT + 255) / 256)), t256, 0, stream>>>(out, f1, alpha, ACT);
}
